// CANN_8469675508085
// MI455X (gfx1250) — matrix operand order repaired and checked by reading, not compiled, not run on hardware
//
#include <hip/hip_runtime.h>

typedef __attribute__((ext_vector_type(16))) _Float16 v16h;
typedef __attribute__((ext_vector_type(8)))  _Float16 v8h;
typedef __attribute__((ext_vector_type(8)))  float    v8f;

#define CH 16
#define HW 65536
#define WPX 256
#define THRESH 0.1f
#define URATE 0.25f

__device__ __forceinline__ void lds_fence() {
  asm volatile("s_wait_dscnt 0" ::: "memory");
}

__launch_bounds__(256)
__global__ void nca_update_kernel(const float* __restrict__ xin,
                                  const float* __restrict__ w1,
                                  const float* __restrict__ b1,
                                  const float* __restrict__ w2,
                                  const float* __restrict__ b2,
                                  const float* __restrict__ w3,
                                  const float* __restrict__ um,
                                  float* __restrict__ xmid,
                                  float* __restrict__ preg) {
  __shared__ float xs[CH * 324];                       // 16 ch x 18x18 halo tile
  __shared__ __align__(32) _Float16 wz1[4096];         // L1 B-frags [kt*4+nt][lane][e]
  __shared__ __align__(32) _Float16 wz2[4096];         // L2 B-frags
  __shared__ __align__(32) _Float16 wz3[1024];         // L3 B-frags [kt][lane][e]
  __shared__ float b1L[64];
  __shared__ float b2L[64];
  // per-wave scratch, unioned in time:
  //   phase A: fz[p][K]  row-major features, stride 72 f16 (16B-aligned chunks)
  //   phase B: hswz[kt*512 + la*16 + e]  A-swizzled hidden state
  __shared__ __align__(32) _Float16 scr[8][1152];
  __shared__ float umL[8][16];                         // per-wave update mask row

  const int t    = threadIdx.x;
  const int b    = blockIdx.x >> 8;
  const int tile = blockIdx.x & 255;
  const int ty0  = (tile >> 4) << 4;
  const int tx0  = (tile & 15) << 4;

  // ---- phase 0: stage x tile + wrap halo into LDS ----
  for (int idx = t; idx < CH * 324; idx += 256) {
    int c   = idx / 324;
    int rem = idx - c * 324;
    int row = rem / 18, col = rem - row * 18;
    int gy = (ty0 + row - 1) & 255;
    int gx = (tx0 + col - 1) & 255;
    xs[idx] = xin[(size_t)(b * CH + c) * HW + gy * WPX + gx];
  }
  // ---- stage weights pre-swizzled into B-fragment lane layout ----
  // B layout: lane l holds N = l%16, element e holds K = kbase + (e<8?e:e+8) + 8*(l/16)
  for (int idx = t; idx < 4096; idx += 256) {
    int frag = idx >> 9;          // kt*4 + nt
    int la   = (idx >> 4) & 31;
    int e    = idx & 15;
    int kt = frag >> 2, nt = frag & 3;
    int k = kt * 32 + (e < 8 ? e : e + 8) + 8 * (la >> 4);
    int n = nt * 16 + (la & 15);
    wz1[idx] = (k < 48) ? (_Float16)w1[n * 48 + k] : (_Float16)0.0f;
    wz2[idx] = (_Float16)w2[n * 64 + k];
  }
  for (int idx = t; idx < 1024; idx += 256) {
    int kt = idx >> 9;
    int la = (idx >> 4) & 31;
    int e  = idx & 15;
    int k = kt * 32 + (e < 8 ? e : e + 8) + 8 * (la >> 4);
    int n = la & 15;
    wz3[idx] = (_Float16)w3[n * 64 + k];
  }
  if (t < 64)       b1L[t]      = b1[t];
  else if (t < 128) b2L[t - 64] = b2[t - 64];
  __syncthreads();

  const int wave = t >> 5;
  const int lane = t & 31;
  const int lh   = lane >> 4;    // lane half
  const int ln   = lane & 15;
  _Float16* fz = &scr[wave][0];

  for (int mi = 0; mi < 2; ++mi) {
    const int r  = wave * 2 + mi;    // pixel row within tile (M-tile)
    const int py = ty0 + r;

    // pre-mask (alive on OLD x) + update mask for this row (lanes 0..15)
    if (lane < 16) {
      const float* ap = xs + 3 * 324 + r * 18 + lane;
      float m = ap[0];
      m = fmaxf(m, ap[1]);  m = fmaxf(m, ap[2]);
      m = fmaxf(m, ap[18]); m = fmaxf(m, ap[19]); m = fmaxf(m, ap[20]);
      m = fmaxf(m, ap[36]); m = fmaxf(m, ap[37]); m = fmaxf(m, ap[38]);
      float pm = (m > THRESH) ? 1.0f : 0.0f;
      preg[(size_t)b * HW + py * WPX + tx0 + lane] = pm;
      float u = um[(size_t)b * HW + py * WPX + tx0 + lane];
      umL[wave][lane] = (u < URATE) ? 1.0f : 0.0f;
    }

    // ---- cooperative perceive: each lane does 8 stencils (p=ln, c=2j+lh),
    //      identity/Gx/Gy share one 9-point load ----
#pragma unroll
    for (int j = 0; j < 8; ++j) {
      int c = 2 * j + lh;
      const float* q = xs + c * 324 + r * 18 + ln;   // top-left of 3x3
      float t00 = q[0],  t01 = q[1],  t02 = q[2];
      float t10 = q[18], t11 = q[19], t12 = q[20];
      float t20 = q[36], t21 = q[37], t22 = q[38];
      float gx = (t02 - t00) + 2.0f * (t12 - t10) + (t22 - t20);
      float gy = (t20 - t00) + 2.0f * (t21 - t01) + (t22 - t02);
      _Float16* d = fz + ln * 72 + 3 * c;
      d[0] = (_Float16)t11;
      d[1] = (_Float16)gx;
      d[2] = (_Float16)gy;
    }
    // zero-pad K = 48..63
#pragma unroll
    for (int j = 0; j < 8; ++j)
      fz[ln * 72 + 48 + 8 * lh + j] = (_Float16)0.0f;
    lds_fence();

    // ---- assemble layer-1 A fragments from fz with aligned 16B vector loads.
    // A layout: lane l holds M = l%16, element e -> K = (e<8?e:e+8) + 8*(l/16)
    v16h a1_0, a1_1;
    {
      const int rb = ln * 72 + 8 * lh;        // 16B-aligned
      v8h c00 = *(const v8h*)&fz[rb];         // kt=0, K = 8lh + 0..7
      v8h c01 = *(const v8h*)&fz[rb + 16];    // kt=0, K = 8lh + 16..23
      v8h c10 = *(const v8h*)&fz[rb + 32];    // kt=1
      v8h c11 = *(const v8h*)&fz[rb + 48];
#pragma unroll
      for (int e = 0; e < 8; ++e) {
        a1_0[e] = c00[e]; a1_0[e + 8] = c01[e];
        a1_1[e] = c10[e]; a1_1[e + 8] = c11[e];
      }
    }

    // ---- layer 1: 48(->64) -> 64, bias+relu, store A-swizzled h1 ----
#pragma unroll
    for (int nt = 0; nt < 4; ++nt) {
      v8f acc = {};
      v16h bf0 = *(const v16h*)&wz1[nt * 512 + lane * 16];
      v16h bf1 = *(const v16h*)&wz1[(4 + nt) * 512 + lane * 16];
      acc = __builtin_amdgcn_wmma_f32_16x16x32_f16(false, a1_0, false, bf0,
                                                   (short)0, acc, false, false);
      acc = __builtin_amdgcn_wmma_f32_16x16x32_f16(false, a1_1, false, bf1,
                                                   (short)0, acc, false, false);
      int n = nt * 16 + ln;          // hidden index == next-layer K
      float bias = b1L[n];
      int kt_ = n >> 5, kl = n & 31;
      int e_, hf;
      if (kl < 8)       { e_ = kl;      hf = 0; }
      else if (kl < 16) { e_ = kl - 8;  hf = 1; }
      else if (kl < 24) { e_ = kl - 8;  hf = 0; }
      else              { e_ = kl - 16; hf = 1; }
#pragma unroll
      for (int v = 0; v < 8; ++v) {
        float h = acc[v] + bias;
        h = h > 0.0f ? h : 0.0f;
        int la = v + 8 * lh + 16 * hf;          // D: M = v + 8*(lane/16)
        scr[wave][kt_ * 512 + la * 16 + e_] = (_Float16)h;
      }
    }
    lds_fence();
    v16h a2_0 = *(const v16h*)&scr[wave][lane * 16];
    v16h a2_1 = *(const v16h*)&scr[wave][512 + lane * 16];

    // ---- layer 2: 64 -> 64, bias+relu, store A-swizzled h2 ----
#pragma unroll
    for (int nt = 0; nt < 4; ++nt) {
      v8f acc = {};
      v16h bf0 = *(const v16h*)&wz2[nt * 512 + lane * 16];
      v16h bf1 = *(const v16h*)&wz2[(4 + nt) * 512 + lane * 16];
      acc = __builtin_amdgcn_wmma_f32_16x16x32_f16(false, a2_0, false, bf0,
                                                   (short)0, acc, false, false);
      acc = __builtin_amdgcn_wmma_f32_16x16x32_f16(false, a2_1, false, bf1,
                                                   (short)0, acc, false, false);
      int n = nt * 16 + ln;
      float bias = b2L[n];
      int kt_ = n >> 5, kl = n & 31;
      int e_, hf;
      if (kl < 8)       { e_ = kl;      hf = 0; }
      else if (kl < 16) { e_ = kl - 8;  hf = 1; }
      else if (kl < 24) { e_ = kl - 8;  hf = 0; }
      else              { e_ = kl - 16; hf = 1; }
#pragma unroll
      for (int v = 0; v < 8; ++v) {
        float h = acc[v] + bias;
        h = h > 0.0f ? h : 0.0f;
        int la = v + 8 * lh + 16 * hf;
        scr[wave][kt_ * 512 + la * 16 + e_] = (_Float16)h;
      }
    }
    lds_fence();
    v16h a3_0 = *(const v16h*)&scr[wave][lane * 16];
    v16h a3_1 = *(const v16h*)&scr[wave][512 + lane * 16];

    // ---- layer 3: 64 -> 16 (no bias) ----
    v8f y = {};
    {
      v16h cf0 = *(const v16h*)&wz3[lane * 16];
      v16h cf1 = *(const v16h*)&wz3[512 + lane * 16];
      y = __builtin_amdgcn_wmma_f32_16x16x32_f16(false, a3_0, false, cf0,
                                                 (short)0, y, false, false);
      y = __builtin_amdgcn_wmma_f32_16x16x32_f16(false, a3_1, false, cf1,
                                                 (short)0, y, false, false);
    }

    // ---- x_mid = x + y * um ;  D layout: lane -> channel ln, v -> pixel ----
#pragma unroll
    for (int v = 0; v < 8; ++v) {
      int p = v + 8 * lh;
      float val = xs[ln * 324 + (r + 1) * 18 + (p + 1)] + y[v] * umL[wave][p];
      xmid[(size_t)(b * CH + ln) * HW + py * WPX + tx0 + p] = val;
    }
  }
}

__launch_bounds__(256)
__global__ void nca_mask_kernel(const float* __restrict__ xmid,
                                const float* __restrict__ preg,
                                float* __restrict__ xout) {
  __shared__ float al[324];
  const int t    = threadIdx.x;
  const int b    = blockIdx.x >> 8;
  const int tile = blockIdx.x & 255;
  const int ty0  = (tile >> 4) << 4;
  const int tx0  = (tile & 15) << 4;

  for (int idx = t; idx < 324; idx += 256) {
    int row = idx / 18, col = idx - row * 18;
    int gy = (ty0 + row - 1) & 255;
    int gx = (tx0 + col - 1) & 255;
    al[idx] = xmid[(size_t)(b * CH + 3) * HW + gy * WPX + gx];
  }
  __syncthreads();

  const int px = t & 15, r = t >> 4;
  const float* p = al + r * 18 + px;
  float m = p[0];
  m = fmaxf(m, p[1]);  m = fmaxf(m, p[2]);
  m = fmaxf(m, p[18]); m = fmaxf(m, p[19]); m = fmaxf(m, p[20]);
  m = fmaxf(m, p[36]); m = fmaxf(m, p[37]); m = fmaxf(m, p[38]);
  float post = (m > THRESH) ? 1.0f : 0.0f;

  const int pix = (ty0 + r) * WPX + tx0 + px;
  float mask = post * preg[(size_t)b * HW + pix];
#pragma unroll
  for (int c = 0; c < CH; ++c) {
    size_t o = (size_t)(b * CH + c) * HW + pix;
    xout[o] = xmid[o] * mask;
  }
}

extern "C" void kernel_launch(void* const* d_in, const int* in_sizes, int n_in,
                              void* d_out, int out_size, void* d_ws, size_t ws_size,
                              hipStream_t stream) {
  const float* x  = (const float*)d_in[0];
  const float* w1 = (const float*)d_in[1];
  const float* b1 = (const float*)d_in[2];
  const float* w2 = (const float*)d_in[3];
  const float* b2 = (const float*)d_in[4];
  const float* w3 = (const float*)d_in[5];
  const float* um = (const float*)d_in[6];
  float* out  = (float*)d_out;
  float* xmid = (float*)d_ws;                       // 8*16*256*256 f32 = 33.5 MB
  float* preg = xmid + (size_t)8 * CH * HW;         // 8*256*256 f32 = 2 MB

  dim3 grid(2048), blk(256);
  for (int s = 0; s < 2; ++s) {
    const float* xin = (s == 0) ? x : out;
    nca_update_kernel<<<grid, blk, 0, stream>>>(xin, w1, b1, w2, b2, w3,
                                                um + (size_t)s * 8 * HW,
                                                xmid, preg);
    nca_mask_kernel<<<grid, blk, 0, stream>>>(xmid, preg, out);
  }
}
